// SGMoE_21388937134612
// MI455X (gfx1250) — matrix operand order repaired and checked by reading, not compiled, not run on hardware
//
#include <hip/hip_runtime.h>
#include <hip/hip_bf16.h>
#include <math.h>
#include <stdint.h>

// ---------------------------------------------------------------------------
// Problem constants (from reference setup_inputs)
// ---------------------------------------------------------------------------
constexpr int N_ = 8192;
constexpr int D_ = 512;
constexpr int H_ = 1024;
constexpr int O_ = 512;
constexpr int E_ = 8;

typedef __attribute__((ext_vector_type(16))) _Float16 v16h;
typedef __attribute__((ext_vector_type(8)))  _Float16 v8h;
typedef __attribute__((ext_vector_type(8)))  float    v8f;
typedef uint32_t u32x4 __attribute__((ext_vector_type(4)));
typedef uint32_t u32x8 __attribute__((ext_vector_type(8)));

// ---------------------------------------------------------------------------
// WMMA helpers (gfx1250: D = A(16x32 f16) * B(32x16 f16) + C(16x16 f32))
// ---------------------------------------------------------------------------
static __device__ inline v8f wmma16(v16h a, v16h b, v8f c) {
    return __builtin_amdgcn_wmma_f32_16x16x32_f16(
        /*neg_a=*/false, a, /*neg_b=*/false, b,
        /*c_mod=*/(short)0, c, /*reuse_a=*/false, /*reuse_b=*/false);
}

// A-fragment (16x32 f16). Lanes 0-15: VGPR0-3 K=0..7, VGPR4-7 K=16..23;
// lanes 16-31: K=8..15 / K=24..31.
// => element e maps to K = (e>>3)*16 + (lane>>4)*8 + (e&7): two 16B chunks.
static __device__ inline v16h load_a_frag(const _Float16* rowk0 /* &A[r][k0] */, int hl) {
    const _Float16* q = rowk0 + hl * 8;
    v8h lo = *(const v8h*)(q);
    v8h hi = *(const v8h*)(q + 16);
    v16h a;
#pragma unroll
    for (int i = 0; i < 8; ++i) { a[i] = lo[i]; a[8 + i] = hi[i]; }
    return a;
}

// B-fragment (32x16 f16), B stored column-major-per-output (W[col][k]):
// lane holds column (lane&15), K = (e>>3)*16 + (lane>>4)*8 + (e&7): two 16B chunks.
static __device__ inline v16h load_b_frag(const _Float16* colbase, int k0, int hl) {
    const _Float16* q = colbase + k0 + hl * 8;
    v8h lo = *(const v8h*)(q);
    v8h hi = *(const v8h*)(q + 16);
    v16h b;
#pragma unroll
    for (int i = 0; i < 8; ++i) { b[i] = lo[i]; b[8 + i] = hi[i]; }
    return b;
}

// Async global->LDS copy of one 16B chunk (per-lane addresses), ASYNCcnt-tracked.
static __device__ inline void async_copy_b128(const void* gsrc, void* lds_dst) {
    uint32_t lds_off = (uint32_t)(uintptr_t)lds_dst;   // LDS aperture: low 32 bits
    asm volatile("global_load_async_to_lds_b128 %0, %1, off"
                 :: "v"(lds_off), "v"(gsrc) : "memory");
}
static __device__ inline void async_wait0() {
    asm volatile("s_wait_asynccnt 0x0" ::: "memory");
}

// ---------------------------------------------------------------------------
// Small utility kernels
// ---------------------------------------------------------------------------
__global__ void zero_ctrl_kernel(unsigned int* p) {
    if (threadIdx.x < 32) p[threadIdx.x] = 0u;
}

__global__ void cvt_f16_kernel(const float* __restrict__ s, _Float16* __restrict__ d, int n) {
    for (int i = blockIdx.x * blockDim.x + threadIdx.x; i < n; i += gridDim.x * blockDim.x)
        d[i] = (_Float16)s[i];
}

// output_weight is (D, O) row-major; store transposed f16: Wt[o*D + d]
__global__ void transpose_f16_kernel(const float* __restrict__ s, _Float16* __restrict__ d) {
    int n = D_ * O_;
    for (int i = blockIdx.x * blockDim.x + threadIdx.x; i < n; i += gridDim.x * blockDim.x) {
        int dd = i >> 9;      // / O_
        int oo = i & (O_ - 1);
        d[oo * D_ + dd] = (_Float16)s[i];
    }
}

// ---------------------------------------------------------------------------
// Gating kernel: layernorm + noisy top-k gating + routing lists + loss sums
// ---------------------------------------------------------------------------
__global__ __launch_bounds__(256) void gating_kernel(
    const float* __restrict__ x, const float* __restrict__ noise,
    const float* __restrict__ nw, const float* __restrict__ nb,
    const float* __restrict__ wg, const float* __restrict__ wn,
    _Float16* __restrict__ xh_norm,
    int* __restrict__ expert_count, float* __restrict__ gates_colsum,
    float* __restrict__ load_colsum, float* __restrict__ lengths,
    int* __restrict__ expert_rows, float* __restrict__ expert_gate)
{
    __shared__ float sx[D_];
    __shared__ float red[256];
    __shared__ float sc[E_], snz[E_];

    const int tid = threadIdx.x;
    const int row = blockIdx.x;
    const size_t rb = (size_t)row * D_;

    float v0 = x[rb + tid];
    float v1 = x[rb + tid + 256];

    red[tid] = v0 + v1;
    __syncthreads();
    for (int s = 128; s > 0; s >>= 1) { if (tid < s) red[tid] += red[tid + s]; __syncthreads(); }
    float mu = red[0] * (1.0f / D_);
    __syncthreads();

    float d0 = v0 - mu, d1 = v1 - mu;
    red[tid] = d0 * d0 + d1 * d1;
    __syncthreads();
    for (int s = 128; s > 0; s >>= 1) { if (tid < s) red[tid] += red[tid + s]; __syncthreads(); }
    float var = red[0] * (1.0f / D_);
    float rs = rsqrtf(var + 1e-6f);

    float xn0 = d0 * rs * nw[tid]       + nb[tid];
    float xn1 = d1 * rs * nw[tid + 256] + nb[tid + 256];
    sx[tid] = xn0; sx[tid + 256] = xn1;
    xh_norm[rb + tid]       = (_Float16)xn0;
    xh_norm[rb + tid + 256] = (_Float16)xn1;
    __syncthreads();

    // gate / noise logits: wave w handles expert w (8 waves of 32 lanes)
    const int lane = tid & 31;
    const int w = tid >> 5;
    float cg = 0.f, cn = 0.f;
    for (int d = lane; d < D_; d += 32) {
        float xv = sx[d];
        cg = fmaf(xv, wg[d * E_ + w], cg);
        cn = fmaf(xv, wn[d * E_ + w], cn);
    }
#pragma unroll
    for (int off = 16; off > 0; off >>= 1) {
        cg += __shfl_down(cg, off, 32);
        cn += __shfl_down(cn, off, 32);
    }
    if (lane == 0) { sc[w] = cg; snz[w] = cn; }
    __syncthreads();

    if (tid == 0) {
        float clean[E_], stdv[E_], nz[E_];
#pragma unroll
        for (int e = 0; e < E_; ++e) {
            float r = snz[e];
            float sp = (r > 20.f) ? r : log1pf(expf(r));     // softplus
            stdv[e]  = sp + 0.01f;                           // NOISE_EPS
            clean[e] = sc[e];
            nz[e]    = clean[e] + noise[(size_t)row * E_ + e] * stdv[e];
        }
        // top-3 of 8 (lowest index wins ties, like lax.top_k)
        bool used[E_] = {false,false,false,false,false,false,false,false};
        int   idx[3]; float val[3];
#pragma unroll
        for (int j = 0; j < 3; ++j) {
            float best = -INFINITY; int bi = 0;
#pragma unroll
            for (int e = 0; e < E_; ++e)
                if (!used[e] && nz[e] > best) { best = nz[e]; bi = e; }
            idx[j] = bi; val[j] = best; used[bi] = true;
        }
        // softmax over top-2
        float e1 = expf(val[1] - val[0]);
        float g0 = 1.0f / (1.0f + e1);
        float g1 = e1 * g0;

        atomicAdd(&gates_colsum[idx[0]], g0);
        atomicAdd(&gates_colsum[idx[1]], g1);
        atomicAdd(&lengths[idx[0]], 1.0f);
        atomicAdd(&lengths[idx[1]], 1.0f);

        int p0 = atomicAdd(&expert_count[idx[0]], 1);
        expert_rows[idx[0] * N_ + p0] = row;
        expert_gate[idx[0] * N_ + p0] = g0;
        int p1 = atomicAdd(&expert_count[idx[1]], 1);
        expert_rows[idx[1] * N_ + p1] = row;
        expert_gate[idx[1] * N_ + p1] = g1;

        // load probs
        const float thr_in = val[2], thr_out = val[1];
        const float inv_sqrt2 = 0.70710678118654752f;
#pragma unroll
        for (int e = 0; e < E_; ++e) {
            float thr = (nz[e] > thr_in) ? thr_in : thr_out;
            float z = (clean[e] - thr) / stdv[e];
            float p = 0.5f * (1.0f + erff(z * inv_sqrt2));
            atomicAdd(&load_colsum[e], p);
        }
    }
}

// ---------------------------------------------------------------------------
// Dense output projection: y = x @ output_weight  (M=N_, K=D_, N=O_)
// 512 blocks x 512 threads (16 waves); each wave: 16 rows x 32 cols.
// x tile staged to LDS with async global->LDS (ASYNCcnt) instructions.
// ---------------------------------------------------------------------------
__global__ __launch_bounds__(512) void proj_kernel(
    const _Float16* __restrict__ xh,     // x in f16, N_ x D_
    const _Float16* __restrict__ Wt,     // output_weight^T f16, O_ x D_
    float* __restrict__ y)
{
    __shared__ _Float16 As[16][D_ + 8];

    const int tid  = threadIdx.x;
    const int tile = blockIdx.x;        // 16-row tile

    // async-stage 16 rows of x (f16) into LDS, 16B chunks per lane
    for (int c = tid; c < 16 * (D_ / 8); c += 512) {
        int r = c >> 6;                 // D_/8 = 64 chunks per row
        int off = (c & 63) * 8;
        async_copy_b128(&xh[((size_t)(tile * 16 + r)) * D_ + off], &As[r][off]);
    }
    async_wait0();
    __syncthreads();

    const int lane = tid & 31, wave = tid >> 5;
    const int hl = lane >> 4, nn = lane & 15;
    const int colbase = wave * 32;

    v8f zero8 = {0.f,0.f,0.f,0.f,0.f,0.f,0.f,0.f};
    v8f acc[2] = {zero8, zero8};
    const _Float16* bp0 = Wt + (size_t)(colbase +  0 + nn) * D_;
    const _Float16* bp1 = Wt + (size_t)(colbase + 16 + nn) * D_;

    for (int k0 = 0; k0 < D_; k0 += 32) {
        v16h a = load_a_frag(&As[nn][k0], hl);
        acc[0] = wmma16(a, load_b_frag(bp0, k0, hl), acc[0]);
        acc[1] = wmma16(a, load_b_frag(bp1, k0, hl), acc[1]);
    }

#pragma unroll
    for (int t = 0; t < 2; ++t) {
        int col = colbase + t * 16 + nn;
#pragma unroll
        for (int i = 0; i < 8; ++i) {
            int r = i + 8 * hl;
            y[((size_t)(tile * 16 + r)) * O_ + col] = acc[t][i];
        }
    }
}

// ---------------------------------------------------------------------------
// Grouped expert FFN. The 16-token gather is done by the Tensor Data Mover
// in gather mode: 16x 16-bit row indices, tile_dim0 = 512 halves per row,
// LDS padding 4 DWORDs per 256 DWORDs reproduces the [16][520] layout.
// ---------------------------------------------------------------------------
__global__ __launch_bounds__(512) void ffn_kernel(
    const _Float16* __restrict__ xh_norm,            // N_ x D_
    const _Float16* __restrict__ W1h,                // E_ x H_ x D_
    const _Float16* __restrict__ W2h,                // E_ x O_ x H_
    const float* __restrict__ b1, const float* __restrict__ b2,
    const int* __restrict__ expert_count,
    const int* __restrict__ expert_rows, const float* __restrict__ expert_gate,
    float* __restrict__ y)
{
    const int e    = blockIdx.y;
    const int tile = blockIdx.x;
    const int cnt  = expert_count[e];
    if (tile * 16 >= cnt) return;

    __shared__ _Float16 As[16][D_ + 8];
    __shared__ _Float16 Act[16][H_ + 8];
    __shared__ int   srow[16];
    __shared__ float sgate[16];

    const int tid  = threadIdx.x;
    const int lane = tid & 31, wave = tid >> 5;

    if (tid < 16) {
        int slot = tile * 16 + tid;
        if (slot < cnt) {
            srow[tid]  = expert_rows[e * N_ + slot];
            sgate[tid] = expert_gate[e * N_ + slot];
        } else {
            srow[tid] = 0; sgate[tid] = 0.0f;   // padded rows contribute 0
        }
    }

    // ---- TDM gather: 16 token rows of x_norm (f16) -> LDS tile ----
    if (wave == 0) {
        uint32_t idx[16];
#pragma unroll
        for (int i = 0; i < 16; ++i) {
            int slot = tile * 16 + i;
            idx[i] = (slot < cnt) ? (uint32_t)expert_rows[e * N_ + slot] : 0u;
        }
        uint64_t gaddr = (uint64_t)(uintptr_t)xh_norm;
        uint32_t laddr = (uint32_t)(uintptr_t)&As[0][0];

        u32x4 g0;
        g0[0] = 1u | (1u << 31);                       // count=1, gather_mode, 16b idx
        g0[1] = laddr;                                 // lds_addr
        g0[2] = (uint32_t)gaddr;                       // global_addr[31:0]
        g0[3] = (uint32_t)((gaddr >> 32) & 0x01FFFFFFu) | (2u << 30); // [56:32], type=2

        u32x8 g1;
        g1[0] = (1u << 16)                             // data_size = 2 bytes
              | (1u << 20)                             // pad_enable
              | (7u << 22)                             // pad_interval: 256 DWORDs
              | (3u << 25);                            // pad_amount:   4 DWORDs
        g1[1] = ((uint32_t)D_ & 0xFFFFu) << 16;        // tensor_dim0 lo @ [63:48]
        g1[2] = ((uint32_t)D_ >> 16) | (((uint32_t)N_ & 0xFFFFu) << 16); // dim0 hi | dim1 lo
        g1[3] = ((uint32_t)N_ >> 16) | ((uint32_t)D_ << 16);             // dim1 hi | tile_dim0
        g1[4] = 16u;                                   // tile_dim1 = #valid indices
        g1[5] = (uint32_t)D_;                          // tensor_dim0_stride[31:0]
        g1[6] = 0u;                                    // stride hi | dim1_stride lo (ignored)
        g1[7] = 0u;

        u32x4 g2, g3;
#pragma unroll
        for (int i = 0; i < 4; ++i) {
            g2[i] = idx[2 * i]     | (idx[2 * i + 1] << 16);
            g3[i] = idx[8 + 2 * i] | (idx[8 + 2 * i + 1] << 16);
        }
        asm volatile("tensor_load_to_lds %0, %1, %2, %3"
                     :: "s"(g0), "s"(g1), "s"(g2), "s"(g3) : "memory");
        asm volatile("s_wait_tensorcnt 0x0" ::: "memory");
    }
    __syncthreads();

    const int hl = lane >> 4, nn = lane & 15;
    v8f zero8 = {0.f,0.f,0.f,0.f,0.f,0.f,0.f,0.f};

    // ---- GEMM1: h(16 x H_) = As(16 x D_) * W1[e]^T ----
    {
        const _Float16* W1e = W1h + (size_t)e * H_ * D_;
        const int colbase = wave * 64;
        v8f acc[4] = {zero8, zero8, zero8, zero8};
        const _Float16* bp[4];
#pragma unroll
        for (int t = 0; t < 4; ++t)
            bp[t] = W1e + (size_t)(colbase + t * 16 + nn) * D_;

        for (int k0 = 0; k0 < D_; k0 += 32) {
            v16h a = load_a_frag(&As[nn][k0], hl);
#pragma unroll
            for (int t = 0; t < 4; ++t)
                acc[t] = wmma16(a, load_b_frag(bp[t], k0, hl), acc[t]);
        }
#pragma unroll
        for (int t = 0; t < 4; ++t) {
            int col = colbase + t * 16 + nn;
            float bias = b1[e * H_ + col];
#pragma unroll
            for (int i = 0; i < 8; ++i) {
                int r = i + 8 * hl;
                float h = acc[t][i] + bias;
                float a = h / (1.0f + expf(-h));            // silu
                Act[r][col] = (_Float16)a;
            }
        }
    }
    __syncthreads();

    // ---- GEMM2: ye(16 x O_) = Act(16 x H_) * W2[e]^T ; scatter-add ----
    {
        const _Float16* W2e = W2h + (size_t)e * O_ * H_;
        const int colbase = wave * 32;
        v8f acc[2] = {zero8, zero8};
        const _Float16* bp0 = W2e + (size_t)(colbase +  0 + nn) * H_;
        const _Float16* bp1 = W2e + (size_t)(colbase + 16 + nn) * H_;

        for (int k0 = 0; k0 < H_; k0 += 32) {
            v16h a = load_a_frag(&Act[nn][k0], hl);
            acc[0] = wmma16(a, load_b_frag(bp0, k0, hl), acc[0]);
            acc[1] = wmma16(a, load_b_frag(bp1, k0, hl), acc[1]);
        }
#pragma unroll
        for (int t = 0; t < 2; ++t) {
            int col = colbase + t * 16 + nn;
            float bias = b2[e * O_ + col];
#pragma unroll
            for (int i = 0; i < 8; ++i) {
                int r = i + 8 * hl;
                float val = (acc[t][i] + bias) * sgate[r];
                atomicAdd(&y[(size_t)srow[r] * O_ + col], val);
            }
        }
    }
}

// ---------------------------------------------------------------------------
// Finalize: loss scalar + load vector
// ---------------------------------------------------------------------------
__global__ void finalize_kernel(const float* __restrict__ gates_colsum,
                                const float* __restrict__ load_colsum,
                                const float* __restrict__ lengths,
                                float* __restrict__ out /* d_out */)
{
    if (threadIdx.x != 0 || blockIdx.x != 0) return;
    float s1 = 0.f, q1 = 0.f, s2 = 0.f, q2 = 0.f, sl = 0.f;
#pragma unroll
    for (int e = 0; e < E_; ++e) {
        float g = gates_colsum[e]; s1 += g; q1 += g * g;
        float l = load_colsum[e];  s2 += l; q2 += l * l;
        sl += lengths[e];
    }
    float m1 = s1 / E_, v1 = q1 / E_ - m1 * m1;
    float m2 = s2 / E_, v2 = q2 / E_ - m2 * m2;
    float cv1 = v1 / (m1 * m1 + 1e-10f);
    float cv2 = v2 / (m2 * m2 + 1e-10f);
    size_t base = (size_t)N_ * O_;
    out[base] = 0.01f * (cv1 + cv2);                     // LOSS_COEF
#pragma unroll
    for (int e = 0; e < E_; ++e)
        out[base + 1 + e] = lengths[e] / sl;
}

// ---------------------------------------------------------------------------
// Launch orchestration
// ---------------------------------------------------------------------------
extern "C" void kernel_launch(void* const* d_in, const int* in_sizes, int n_in,
                              void* d_out, int out_size, void* d_ws, size_t ws_size,
                              hipStream_t stream) {
    (void)in_sizes; (void)n_in; (void)out_size; (void)ws_size;

    const float* x      = (const float*)d_in[0];
    const float* noise  = (const float*)d_in[3];
    const float* norm_w = (const float*)d_in[4];
    const float* norm_b = (const float*)d_in[5];
    const float* w_gate = (const float*)d_in[6];
    const float* w_nse  = (const float*)d_in[7];
    const float* W1     = (const float*)d_in[8];
    const float* b1     = (const float*)d_in[9];
    const float* W2     = (const float*)d_in[10];
    const float* b2     = (const float*)d_in[11];
    const float* wout   = (const float*)d_in[12];

    float* y = (float*)d_out;

    // workspace layout (all offsets 256B aligned)
    char* ws = (char*)d_ws;
    size_t off = 0;
    auto take = [&](size_t bytes) { char* p = ws + off; off += (bytes + 255) & ~(size_t)255; return p; };
    _Float16* xh_norm = (_Float16*)take((size_t)N_ * D_ * 2);
    _Float16* xh_raw  = (_Float16*)take((size_t)N_ * D_ * 2);
    _Float16* W1h     = (_Float16*)take((size_t)E_ * H_ * D_ * 2);
    _Float16* W2h     = (_Float16*)take((size_t)E_ * O_ * H_ * 2);
    _Float16* WoutT   = (_Float16*)take((size_t)O_ * D_ * 2);
    int*      erows   = (int*)     take((size_t)E_ * N_ * 4);
    float*    egate   = (float*)   take((size_t)E_ * N_ * 4);
    char*     ctrl    =            take(256);
    int*   expert_count = (int*)  (ctrl + 0);
    float* gates_colsum = (float*)(ctrl + 32);
    float* load_colsum  = (float*)(ctrl + 64);
    float* lengths      = (float*)(ctrl + 96);

    // 1) zero control block (counts + reduction sums) every call
    zero_ctrl_kernel<<<1, 32, 0, stream>>>((unsigned int*)ctrl);

    // 2) f32 -> f16 conversions for WMMA operands
    cvt_f16_kernel<<<2048, 256, 0, stream>>>(x,  xh_raw, N_ * D_);
    cvt_f16_kernel<<<2048, 256, 0, stream>>>(W1, W1h,    E_ * H_ * D_);
    cvt_f16_kernel<<<2048, 256, 0, stream>>>(W2, W2h,    E_ * O_ * H_);
    transpose_f16_kernel<<<512, 256, 0, stream>>>(wout, WoutT);

    // 3) layernorm + gating + routing
    gating_kernel<<<N_, 256, 0, stream>>>(x, noise, norm_w, norm_b, w_gate, w_nse,
                                          xh_norm, expert_count, gates_colsum,
                                          load_colsum, lengths, erows, egate);

    // 4) dense output projection writes y (no prior init needed)
    proj_kernel<<<N_ / 16, 512, 0, stream>>>(xh_raw, WoutT, y);

    // 5) grouped sparse expert FFN, atomically accumulates gate * y_e into y
    dim3 ffn_grid(N_ / 16, E_);
    ffn_kernel<<<ffn_grid, 512, 0, stream>>>(xh_norm, W1h, W2h, b1, b2,
                                             expert_count, erows, egate, y);

    // 6) scalar loss + load vector
    finalize_kernel<<<1, 1, 0, stream>>>(gates_colsum, load_colsum, lengths, y);
}
